// CGCNNPyGChargeEarlySelectivePooling_74637941670367
// MI455X (gfx1250) — matrix operand order repaired and checked by reading, not compiled, not run on hardware
//
#include <hip/hip_runtime.h>
#include <hip/hip_bf16.h>
#include <stddef.h>
#include <stdint.h>

typedef __attribute__((ext_vector_type(16))) _Float16 v16h;
typedef __attribute__((ext_vector_type(8)))  _Float16 v8h;
typedef __attribute__((ext_vector_type(8)))  float    v8f;

// ---------------- constants (match reference) ----------------
constexpr int cN  = 65536;    // nodes
constexpr int cG  = 256;      // graphs
constexpr int cP  = 256;      // nodes per graph
constexpr int cE  = 1048576;  // edges
constexpr int cAF = 92;
constexpr int cBF = 41;
constexpr int cD  = 64;
constexpr int cH  = 128;
constexpr int cCE = 16;
constexpr int cT  = 4;
constexpr float cEPS = 1e-5f;

// ---------------- helpers ----------------
__device__ __forceinline__ float softplusf(float x) {
  return (x > 20.f) ? x : __logf(1.f + __expf(x));
}

// A-fragment (16x32 f16, row = lane&15) from a global f16 row pointer.
// Per-lane data is two contiguous 16B runs: halves [kbase, kbase+8) and
// [16+kbase, 16+kbase+8) with kbase = (lane>=16)*8.
__device__ __forceinline__ v16h load_a_f16g(const _Float16* __restrict__ rowp, int lane) {
  const int kbase = (lane >> 4) << 3;
  v8h lo = *reinterpret_cast<const v8h*>(rowp + kbase);
  v8h hi = *reinterpret_cast<const v8h*>(rowp + 16 + kbase);
  return __builtin_shufflevector(lo, hi, 0, 1, 2, 3, 4, 5, 6, 7,
                                 8, 9, 10, 11, 12, 13, 14, 15);
}

// Same layout from an f16 row in LDS.
__device__ __forceinline__ v16h load_a_lds(const _Float16* rowp, int lane) {
  const int kbase = (lane >> 4) << 3;
  v8h lo = *reinterpret_cast<const v8h*>(rowp + kbase);
  v8h hi = *reinterpret_cast<const v8h*>(rowp + 16 + kbase);
  return __builtin_shufflevector(lo, hi, 0, 1, 2, 3, 4, 5, 6, 7,
                                 8, 9, 10, 11, 12, 13, 14, 15);
}

// B-fragment (32x16 f16) from transposed weights in LDS: Wt[n][k].
// Per-lane: two 8-half runs at k0 + (lane>=16)*8 and k0 + 16 + (lane>=16)*8.
__device__ __forceinline__ v16h load_b_lds(const _Float16* base, int ncol, int stride,
                                           int k0, int lane) {
  const _Float16* p = base + ncol * stride + k0 + ((lane >> 4) << 3);
  v8h lo = *reinterpret_cast<const v8h*>(p);
  v8h hi = *reinterpret_cast<const v8h*>(p + 16);
  return __builtin_shufflevector(lo, hi, 0, 1, 2, 3, 4, 5, 6, 7,
                                 8, 9, 10, 11, 12, 13, 14, 15);
}

__device__ __forceinline__ v8f wmma32(v16h a, v16h b, v8f c) {
  return __builtin_amdgcn_wmma_f32_16x16x32_f16(false, a, false, b, (short)0, c,
                                                false, false);
}

// ---------------- kernel: node embedding init ----------------
__global__ void init_h_kernel(const float* __restrict__ x, const float* __restrict__ charge,
                              const float* __restrict__ cW, const float* __restrict__ cb,
                              const float* __restrict__ aW, const float* __restrict__ ab,
                              const int* __restrict__ batch, float* __restrict__ h,
                              _Float16* __restrict__ h16) {
  __shared__ float feat[cAF + cCE];
  const int n = blockIdx.x;
  const int tid = threadIdx.x; // 64 threads
  for (int k = tid; k < cAF; k += 64) feat[k] = x[(size_t)n * cAF + k];
  if (tid < cCE) {
    const int g = batch[n];
    feat[cAF + tid] = charge[g] * cW[tid] + cb[tid];
  }
  __syncthreads();
  float acc = ab[tid];
#pragma unroll 4
  for (int k = 0; k < cAF + cCE; ++k) acc += feat[k] * aW[k * cD + tid];
  h[(size_t)n * cD + tid] = acc;
  h16[(size_t)n * cD + tid] = (_Float16)acc;
}

// ---------------- kernel: edge embedding init (f16 out) ----------------
__global__ void init_e_kernel(const float* __restrict__ ea, const float* __restrict__ bW,
                              const float* __restrict__ bb, _Float16* __restrict__ e16) {
  __shared__ float feat[cBF];
  const size_t eid = blockIdx.x;
  const int tid = threadIdx.x; // 64 threads
  for (int k = tid; k < cBF; k += 64) feat[k] = ea[eid * cBF + k];
  __syncthreads();
  float acc = bb[tid];
#pragma unroll
  for (int k = 0; k < cBF; ++k) acc += feat[k] * bW[k * cD + tid];
  e16[eid * cD + tid] = (_Float16)acc;
}

// ---------------- kernel: zero hn + stats ----------------
__global__ void zero_kernel(float* __restrict__ hn, float* __restrict__ stats, int n) {
  const int idx = blockIdx.x * blockDim.x + threadIdx.x;
  if (idx < n) hn[idx] = 0.f;
  if (idx < 2 * cD) stats[idx] = 0.f;
}

// ---------------- main WMMA kernel ----------------
constexpr int TPW = 8;
__global__ void __launch_bounds__(128)
edge_msg_kernel(const _Float16* __restrict__ h16, _Float16* __restrict__ e16,
                float* __restrict__ hn, const int* __restrict__ row,
                const int* __restrict__ col,
                const float* __restrict__ euW1, const float* __restrict__ eub1,
                const float* __restrict__ euW2, const float* __restrict__ eub2,
                const float* __restrict__ nuW1, const float* __restrict__ nub1,
                const float* __restrict__ nuW2, const float* __restrict__ nub2) {
  __shared__ __align__(128) _Float16 smem[32768]; // 64 KB
  _Float16* EU1T = smem;           // [64][192] transposed eu_W1
  _Float16* EU2T = smem + 12288;   // [64][64]  transposed eu_W2
  _Float16* NU1T = smem + 16384;   // [64][128] transposed nu_W1
  _Float16* NU2T = smem + 24576;   // [64][64]  transposed nu_W2

  const int tid = threadIdx.x;
  for (int i = tid; i < 64 * 192; i += 128) {
    const int n = i / 192, k = i % 192;
    EU1T[i] = (_Float16)euW1[k * 64 + n];
  }
  for (int i = tid; i < 64 * 64; i += 128) {
    const int n = i / 64, k = i % 64;
    EU2T[i] = (_Float16)euW2[k * 64 + n];
    NU2T[i] = (_Float16)nuW2[k * 64 + n];
  }
  for (int i = tid; i < 64 * 128; i += 128) {
    const int n = i / 128, k = i % 128;
    NU1T[i] = (_Float16)nuW1[k * 64 + n];
  }
  __syncthreads();

  const int wave = tid >> 5;
  const int lane = tid & 31;
  _Float16* ZT = smem + 28672 + wave * 1024; // per-wave [16][64] exchange tile
  const int mrow = lane & 15;
  const int nloc = lane & 15;
  const int hi8  = (lane >> 4) << 3;

  float b_eu1[4], b_eu2[4], b_nu1[4], b_nu2[4];
#pragma unroll
  for (int nt = 0; nt < 4; ++nt) {
    b_eu1[nt] = eub1[nt * 16 + nloc];
    b_eu2[nt] = eub2[nt * 16 + nloc];
    b_nu1[nt] = nub1[nt * 16 + nloc];
    b_nu2[nt] = nub2[nt * 16 + nloc];
  }

  const v8f zc = {}; // zero accumulator -> SRC2 inline 0

  for (int t = 0; t < TPW; ++t) {
    const int tile  = (blockIdx.x * 4 + wave) * TPW + t;
    const int tbase = tile * 16;
    if (tbase >= cE) break;
    const int eidx = tbase + mrow;

    // prefetch next tile's index + edge rows (global_prefetch_b8)
    if (t + 1 < TPW) {
      __builtin_prefetch(row + tbase + 16, 0, 3);
      __builtin_prefetch(col + tbase + 16, 0, 3);
      __builtin_prefetch(e16 + (size_t)(tbase + 16) * cD + mrow * cD, 0, 3);
    }

    const _Float16* hs_row = h16 + (size_t)row[eidx] * cD;
    const _Float16* hd_row = h16 + (size_t)col[eidx] * cD;
    const _Float16* e_row  = e16 + (size_t)eidx * cD;

    // all six A fragments of the edge-update input [hs | hd | e]
    const v16h ahs0 = load_a_f16g(hs_row, lane);
    const v16h ahs1 = load_a_f16g(hs_row + 32, lane);
    const v16h ahd0 = load_a_f16g(hd_row, lane);
    const v16h ahd1 = load_a_f16g(hd_row + 32, lane);
    const v16h aeo0 = load_a_f16g(e_row, lane);
    const v16h aeo1 = load_a_f16g(e_row + 32, lane);

    // ===== edge update, stage 1 =====
    v8f acc[4];
#pragma unroll
    for (int nt = 0; nt < 4; ++nt) {
      const int nc = nt * 16 + nloc;
      acc[nt] = wmma32(ahs0, load_b_lds(EU1T, nc, 192, 0, lane), zc);
      acc[nt] = wmma32(ahs1, load_b_lds(EU1T, nc, 192, 32, lane), acc[nt]);
      acc[nt] = wmma32(ahd0, load_b_lds(EU1T, nc, 192, 64, lane), acc[nt]);
      acc[nt] = wmma32(ahd1, load_b_lds(EU1T, nc, 192, 96, lane), acc[nt]);
      acc[nt] = wmma32(aeo0, load_b_lds(EU1T, nc, 192, 128, lane), acc[nt]);
      acc[nt] = wmma32(aeo1, load_b_lds(EU1T, nc, 192, 160, lane), acc[nt]);
    }
    // bias + softplus, D-layout -> A-layout via per-wave LDS tile
#pragma unroll
    for (int nt = 0; nt < 4; ++nt)
#pragma unroll
      for (int r = 0; r < 8; ++r)
        ZT[(r + hi8) * 64 + nt * 16 + nloc] =
            (_Float16)softplusf(acc[nt][r] + b_eu1[nt]);

    // ===== edge update, stage 2 -> new e =====
    {
      const v16h az0 = load_a_lds(ZT + mrow * 64, lane);
      const v16h az1 = load_a_lds(ZT + mrow * 64 + 32, lane);
#pragma unroll
      for (int nt = 0; nt < 4; ++nt) {
        const int nc = nt * 16 + nloc;
        acc[nt] = wmma32(az0, load_b_lds(EU2T, nc, 64, 0, lane), zc);
        acc[nt] = wmma32(az1, load_b_lds(EU2T, nc, 64, 32, lane), acc[nt]);
      }
    }
    // e_new (f16) into the LDS tile (row-major, matches e16 layout)
#pragma unroll
    for (int nt = 0; nt < 4; ++nt)
#pragma unroll
      for (int r = 0; r < 8; ++r)
        ZT[(r + hi8) * 64 + nt * 16 + nloc] = (_Float16)(acc[nt][r] + b_eu2[nt]);

    // drain e_new tile to global via async LDS->memory DMA (ASYNCcnt path)
    {
      uint64_t gaddr = (uint64_t)(uintptr_t)(e16 + (size_t)tbase * cD) +
                       (uint32_t)(lane * 16);
      uint32_t laddr = (uint32_t)(uintptr_t)ZT + (uint32_t)(lane * 16);
      asm volatile("s_wait_dscnt 0x0" ::: "memory");
      asm volatile("global_store_async_from_lds_b128 %0, %1, off"
                   ::"v"(gaddr), "v"(laddr) : "memory");
      asm volatile("global_store_async_from_lds_b128 %0, %1, off offset:512"
                   ::"v"(gaddr), "v"(laddr) : "memory");
      asm volatile("global_store_async_from_lds_b128 %0, %1, off offset:1024"
                   ::"v"(gaddr), "v"(laddr) : "memory");
      asm volatile("global_store_async_from_lds_b128 %0, %1, off offset:1536"
                   ::"v"(gaddr), "v"(laddr) : "memory");
    }

    // ===== node message, stage 1: [hs | e_new] =====
    {
      const v16h ae0 = load_a_lds(ZT + mrow * 64, lane);
      const v16h ae1 = load_a_lds(ZT + mrow * 64 + 32, lane);
#pragma unroll
      for (int nt = 0; nt < 4; ++nt) {
        const int nc = nt * 16 + nloc;
        acc[nt] = wmma32(ahs0, load_b_lds(NU1T, nc, 128, 0, lane), zc);
        acc[nt] = wmma32(ahs1, load_b_lds(NU1T, nc, 128, 32, lane), acc[nt]);
        acc[nt] = wmma32(ae0, load_b_lds(NU1T, nc, 128, 64, lane), acc[nt]);
        acc[nt] = wmma32(ae1, load_b_lds(NU1T, nc, 128, 96, lane), acc[nt]);
      }
    }
    // wait for the async drain before recycling the tile
    asm volatile("s_wait_asynccnt 0x0" ::: "memory");
#pragma unroll
    for (int nt = 0; nt < 4; ++nt)
#pragma unroll
      for (int r = 0; r < 8; ++r)
        ZT[(r + hi8) * 64 + nt * 16 + nloc] =
            (_Float16)softplusf(acc[nt][r] + b_nu1[nt]);

    // ===== node message, stage 2 -> m =====
    {
      const v16h az0 = load_a_lds(ZT + mrow * 64, lane);
      const v16h az1 = load_a_lds(ZT + mrow * 64 + 32, lane);
#pragma unroll
      for (int nt = 0; nt < 4; ++nt) {
        const int nc = nt * 16 + nloc;
        acc[nt] = wmma32(az0, load_b_lds(NU2T, nc, 64, 0, lane), zc);
        acc[nt] = wmma32(az1, load_b_lds(NU2T, nc, 64, 32, lane), acc[nt]);
      }
    }

    // scatter-add messages: hn[col[edge]] += m + bias
    int cdst[8];
#pragma unroll
    for (int r = 0; r < 8; ++r) cdst[r] = col[tbase + r + hi8];
#pragma unroll
    for (int nt = 0; nt < 4; ++nt)
#pragma unroll
      for (int r = 0; r < 8; ++r)
        atomicAdd(&hn[(size_t)cdst[r] * cD + nt * 16 + nloc],
                  acc[nt][r] + b_nu2[nt]);
  }
}

// ---------------- batchnorm stats ----------------
__global__ void bn_stats_kernel(const float* __restrict__ hn, float* __restrict__ stats) {
  const int gtid = blockIdx.x * blockDim.x + threadIdx.x; // 65536 threads
  const int f = gtid & 63;
  const int chunk = gtid >> 6; // 0..1023
  float s = 0.f, s2 = 0.f;
  for (int r = chunk; r < cN; r += 1024) {
    const float v = hn[(size_t)r * cD + f];
    s += v; s2 += v * v;
  }
  atomicAdd(&stats[f], s);
  atomicAdd(&stats[cD + f], s2);
}

// ---------------- batchnorm apply + softplus + residual ----------------
__global__ void bn_apply_kernel(float* __restrict__ h, _Float16* __restrict__ h16,
                                const float* __restrict__ hn,
                                const float* __restrict__ stats,
                                const float* __restrict__ gamma,
                                const float* __restrict__ beta) {
  const size_t idx = (size_t)blockIdx.x * blockDim.x + threadIdx.x;
  const int f = (int)(idx & 63);
  const float inv_n = 1.f / (float)cN;
  const float mu = stats[f] * inv_n;
  const float var = stats[cD + f] * inv_n - mu * mu;
  const float v = (hn[idx] - mu) * rsqrtf(var + cEPS) * gamma[f] + beta[f];
  const float nh = softplusf(v) + h[idx];
  h[idx] = nh;
  h16[idx] = (_Float16)nh;
}

// ---------------- selective pooling + predictor head ----------------
__global__ void pool_pred_kernel(const float* __restrict__ h, const int* __restrict__ tsi,
                                 const float* __restrict__ W1, const float* __restrict__ b1,
                                 const float* __restrict__ W2, const float* __restrict__ b2,
                                 const float* __restrict__ W3, const float* __restrict__ b3,
                                 float* __restrict__ out) {
  __shared__ float repr[cD];
  __shared__ float z1[cH];
  __shared__ float z2[cH];
  const int g = blockIdx.x;
  const int tid = threadIdx.x; // 128 threads
  if (tid < cD) {
    float s = 0.f;
    int cnt = 0;
#pragma unroll
    for (int t = 0; t < cT; ++t) {
      const int si = tsi[g * cT + t];
      if (si >= 0 && si < cP) {
        s += h[(size_t)(g * cP + si) * cD + tid];
        ++cnt;
      }
    }
    repr[tid] = s / (float)(cnt > 0 ? cnt : 1);
  }
  __syncthreads();
  float a = b1[tid];
#pragma unroll 4
  for (int d = 0; d < cD; ++d) a += repr[d] * W1[d * cH + tid];
  z1[tid] = softplusf(a);
  __syncthreads();
  float c = b2[tid];
#pragma unroll 4
  for (int k = 0; k < cH; ++k) c += z1[k] * W2[k * cH + tid];
  z2[tid] = softplusf(c) * W3[tid];
  __syncthreads();
  for (int s = cH / 2; s > 0; s >>= 1) {
    if (tid < s) z2[tid] += z2[tid + s];
    __syncthreads();
  }
  if (tid == 0) out[g] = z2[0] + b3[0];
}

// ---------------- host launcher ----------------
extern "C" void kernel_launch(void* const* d_in, const int* in_sizes, int n_in,
                              void* d_out, int out_size, void* d_ws, size_t ws_size,
                              hipStream_t stream) {
  const float* x      = (const float*)d_in[0];
  const float* ea     = (const float*)d_in[1];
  const float* charge = (const float*)d_in[2];
  const float* cW     = (const float*)d_in[3];
  const float* cb     = (const float*)d_in[4];
  const float* aW     = (const float*)d_in[5];
  const float* ab     = (const float*)d_in[6];
  const float* bW     = (const float*)d_in[7];
  const float* bb     = (const float*)d_in[8];
  const float* nuW1   = (const float*)d_in[9];
  const float* nub1   = (const float*)d_in[10];
  const float* nuW2   = (const float*)d_in[11];
  const float* nub2   = (const float*)d_in[12];
  const float* euW1   = (const float*)d_in[13];
  const float* eub1   = (const float*)d_in[14];
  const float* euW2   = (const float*)d_in[15];
  const float* eub2   = (const float*)d_in[16];
  const float* gamma  = (const float*)d_in[17];
  const float* beta   = (const float*)d_in[18];
  const float* pW1    = (const float*)d_in[19];
  const float* pb1    = (const float*)d_in[20];
  const float* pW2    = (const float*)d_in[21];
  const float* pb2    = (const float*)d_in[22];
  const float* pW3    = (const float*)d_in[23];
  const float* pb3    = (const float*)d_in[24];
  const int*   eidx   = (const int*)d_in[25];
  const int*   batch  = (const int*)d_in[26];
  const int*   tsi    = (const int*)d_in[27];
  float*       out    = (float*)d_out;

  const int* row = eidx;
  const int* col = eidx + cE;

  char* ws = (char*)d_ws;
  constexpr size_t off_e16 = 0;                                    // E*64*2 = 128 MB
  constexpr size_t off_h   = off_e16 + (size_t)cE * cD * 2;        // N*64*4 = 16 MB
  constexpr size_t off_h16 = off_h + (size_t)cN * cD * 4;          // N*64*2 = 8 MB
  constexpr size_t off_hn  = off_h16 + (size_t)cN * cD * 2;        // N*64*4 = 16 MB
  constexpr size_t off_st  = off_hn + (size_t)cN * cD * 4;

  _Float16* e16_buf = (_Float16*)(ws + off_e16);
  float*    h_buf   = (float*)(ws + off_h);
  _Float16* h16_buf = (_Float16*)(ws + off_h16);
  float*    hn_buf  = (float*)(ws + off_hn);
  float*    stats   = (float*)(ws + off_st);

  init_h_kernel<<<cN, 64, 0, stream>>>(x, charge, cW, cb, aW, ab, batch, h_buf, h16_buf);
  init_e_kernel<<<cE, 64, 0, stream>>>(ea, bW, bb, e16_buf);

  const int edge_blocks = cE / (16 * 4 * TPW); // 2048
  for (int i = 0; i < 3; ++i) {
    zero_kernel<<<(cN * cD + 255) / 256, 256, 0, stream>>>(hn_buf, stats, cN * cD);
    edge_msg_kernel<<<edge_blocks, 128, 0, stream>>>(
        h16_buf, e16_buf, hn_buf, row, col,
        euW1 + (size_t)i * 192 * cD, eub1 + i * cD,
        euW2 + (size_t)i * cD * cD,  eub2 + i * cD,
        nuW1 + (size_t)i * 128 * cD, nub1 + i * cD,
        nuW2 + (size_t)i * cD * cD,  nub2 + i * cD);
    bn_stats_kernel<<<256, 256, 0, stream>>>(hn_buf, stats);
    bn_apply_kernel<<<(cN * cD) / 256, 256, 0, stream>>>(h_buf, h16_buf, hn_buf, stats,
                                                         gamma + i * cD, beta + i * cD);
  }

  pool_pred_kernel<<<cG, cH, 0, stream>>>(h_buf, tsi, pW1, pb1, pW2, pb2, pW3, pb3, out);
}
